// AdaptiveRSNNEmbedding_37074157699161
// MI455X (gfx1250) — matrix operand order repaired and checked by reading, not compiled, not run on hardware
//
#include <hip/hip_runtime.h>
#include <hip/hip_bf16.h>
#include <stdint.h>

// ---------------- problem constants ----------------
#define THRESH  0.5f
#define TS      4
#define NB      16     // time steps
#define BATCH   16
#define CIN     2
#define COUT    32     // gate/current channels (conv emits 2*COUT)
#define HH      160
#define WW      160
#define PW      162    // padded spike plane width/height (halo of 1)

typedef _Float16 half_t;
typedef __attribute__((ext_vector_type(16))) _Float16     v16h;
typedef __attribute__((ext_vector_type(8)))  _Float16     v8h;
typedef __attribute__((ext_vector_type(8)))  float        v8f;
typedef __attribute__((ext_vector_type(8)))  int          v8i;
typedef __attribute__((ext_vector_type(4)))  unsigned int u32x4;
typedef __attribute__((ext_vector_type(4)))  int          i32x4;
typedef __attribute__((ext_vector_type(8)))  int          i32x8;

// ---------------- workspace layout (bytes) ----------------
#define NS        ((size_t)BATCH * COUT * HH * WW)     // 13,107,200 state elems
#define OFF_VMEM  ((size_t)0)
#define OFF_VAVG  (NS * 4)
#define OFF_SEG   (NS * 8)
#define OFF_SPK   (NS * 12)                            // padded f16 spike buffer
#define SPK_ELEMS ((size_t)BATCH * PW * PW * COUT)     // 13,436,928
#define OFF_WG    (OFF_SPK + SPK_ELEMS * 2)            // f16 [9][64][32]
#define OFF_WI    (OFF_WG + (size_t)9 * 64 * 32 * 2)   // f16 [64][32] (K padded)
#define OFF_BIAS  (OFF_WI + (size_t)64 * 32 * 2)       // f32 [64] = b_in + b_gate

#define USE_TDM 1

// ---------------- utility kernels ----------------
__global__ void rsnn_zero(uint32_t* __restrict__ p, size_t n) {
  size_t i = (size_t)blockIdx.x * blockDim.x + threadIdx.x;
  size_t stride = (size_t)gridDim.x * blockDim.x;
  for (; i < n; i += stride) p[i] = 0u;
}

// convert weights to f16 GEMM layouts + combine biases
__global__ void rsnn_prep(const float* __restrict__ wg_f32,   // [64][32][3][3]
                          const float* __restrict__ wi_f32,   // [64][2][3][3]
                          const float* __restrict__ b_in,     // [64]
                          const float* __restrict__ b_gate,   // [64]
                          half_t* __restrict__ wg,            // [9][64][32]
                          half_t* __restrict__ wi,            // [64][32]
                          float*  __restrict__ bias)          // [64]
{
  int tid = blockIdx.x * blockDim.x + threadIdx.x;
  if (tid < 9 * 64 * 32) {
    int tap = tid / (64 * 32);
    int o   = (tid / 32) % 64;
    int i   = tid % 32;
    wg[((size_t)tap * 64 + o) * 32 + i] = (half_t)wg_f32[((size_t)o * 32 + i) * 9 + tap];
  }
  if (tid < 64 * 32) {
    int o = tid / 32, k = tid % 32;
    float v = 0.f;
    if (k < CIN * 9) {                 // K = tap*2 + cin
      int tap = k >> 1, c = k & 1;
      v = wi_f32[((size_t)o * CIN + c) * 9 + tap];
    }
    wi[tid] = (half_t)v;
  }
  if (tid < 64) bias[tid] = b_in[tid] + b_gate[tid];
}

// ---------------- fused time-step kernel ----------------
// block = (batch b, row h); 10 waves, each owns a 16-pixel strip of the row.
__global__ __launch_bounds__(320) void rsnn_step(
    const float*  __restrict__ ev,     // events [B][T][CIN][H][W]
    const half_t* __restrict__ wg,     // [9][64][32]
    const half_t* __restrict__ wi,     // [64][32]
    const float*  __restrict__ bias,   // [64]
    float* __restrict__ vmem,
    float* __restrict__ vavg,
    int*   __restrict__ seg,
    half_t* __restrict__ spk,          // padded [B][PW][PW][COUT]
    float* __restrict__ agg,           // out [TS][B][COUT][H][W]
    int tr)                            // reversed time index into events
{
  __shared__ alignas(16) half_t sp_tile[3 * PW * COUT];  // 31,104 B (pixel-major, ch fastest)
  __shared__ alignas(16) half_t ev_tile[3 * PW * CIN];   //  1,944 B
  __shared__ alignas(16) half_t wg_lds[9 * 64 * 32];     // 36,864 B  [tap][N][K]
  __shared__ alignas(16) half_t wi_lds[64 * 32];         //  4,096 B  [N][K]

  const int b    = blockIdx.x / HH;
  const int h    = blockIdx.x % HH;
  const int tid  = threadIdx.x;
  const int lane = tid & 31;
  const int wave = tid >> 5;          // 0..9
  const int col0 = lane & 15;         // A-matrix row (output pixel within strip)
  const int hi   = lane >> 4;         // half-wave selector for K chunks

  // ---- stage weights into LDS ----
  {
    const uint32_t* s0 = (const uint32_t*)wg;
    uint32_t*       d0 = (uint32_t*)wg_lds;
    for (int i = tid; i < 9 * 64 * 32 / 2; i += 320) d0[i] = s0[i];
    const uint32_t* s1 = (const uint32_t*)wi;
    uint32_t*       d1 = (uint32_t*)wi_lds;
    for (int i = tid; i < 64 * 32 / 2; i += 320) d1[i] = s1[i];
  }

  // ---- stage events halo tile (SAME padding), f32 -> f16 ----
  for (int i = tid; i < 3 * PW * CIN; i += 320) {
    int c   = i % CIN;
    int col = (i / CIN) % PW;
    int row = i / (CIN * PW);
    int gr = h + row - 1;
    int gc = col - 1;
    float v = 0.f;
    if (gr >= 0 && gr < HH && gc >= 0 && gc < WW)
      v = ev[((((size_t)b * NB + tr) * CIN + c) * HH + gr) * WW + gc];
    ev_tile[((size_t)row * PW + col) * CIN + c] = (half_t)v;
  }

  // ---- stage spike halo tile: one contiguous 15,552-half TDM copy ----
#if USE_TDM
  if (wave == 0) {
    const uint32_t NEL   = 3 * PW * COUT;  // 15,552 elements (f16)
    uint64_t gaddr = (uint64_t)(uintptr_t)(spk + ((size_t)b * PW + h) * PW * COUT);
    uint32_t laddr = (uint32_t)(uintptr_t)sp_tile;  // LDS byte offset (low 32 bits of flat)
    u32x4 g0; i32x8 g1;
    i32x4 g2 = {0, 0, 0, 0}, g3 = {0, 0, 0, 0};
    g0[0] = 1u;                                             // count=1, user mode
    g0[1] = laddr;                                          // lds_addr
    g0[2] = (uint32_t)gaddr;                                // global_addr[31:0]
    g0[3] = (uint32_t)((gaddr >> 32) & 0x1FFFFFFu) | (2u << 30);  // addr hi | type=2
    g1[0] = (int)(1u << 16);                                // data_size = 2 bytes
    g1[1] = (int)((NEL & 0xFFFFu) << 16);                   // tensor_dim0[15:0]
    g1[2] = (int)(((NEL >> 16) & 0xFFFFu) | (1u << 16));    // tensor_dim0 hi | tensor_dim1=1
    g1[3] = (int)((NEL & 0xFFFFu) << 16);                   // tile_dim0 (1-D tile)
    g1[4] = 0;                                              // tile_dim1=0, tile_dim2=0
    g1[5] = (int)NEL;                                       // tensor_dim0_stride lo
    g1[6] = 0;
    g1[7] = 0;
#if defined(__clang_major__) && __clang_major__ >= 23
    i32x8 g4 = {0, 0, 0, 0, 0, 0, 0, 0};
    __builtin_amdgcn_tensor_load_to_lds(g0, g1, g2, g3, g4, 0);
#else
    __builtin_amdgcn_tensor_load_to_lds(g0, g1, g2, g3, 0);
#endif
    __builtin_amdgcn_s_wait_tensorcnt(0);
  }
#else
  {
    const half_t* src = spk + ((size_t)b * PW + h) * PW * COUT;
    uint32_t* d = (uint32_t*)sp_tile;
    const uint32_t* s = (const uint32_t*)src;
    for (int i = tid; i < 3 * PW * COUT / 2; i += 320) d[i] = s[i];
  }
#endif
  __syncthreads();

  // ---- implicit GEMM: D(16px x 64ch) = A(16px x K) * B(K x 64ch) ----
  const int w0 = wave * 16;            // strip base (output pixel coords)
  v8f acc0 = {}, acc1 = {}, acc2 = {}, acc3 = {};

  // input conv: K = tap*2 + cin, 18 valid (padded to 32)
  {
    v16h aI;
#pragma unroll
    for (int e = 0; e < 16; ++e) {
      int K = e + hi * 16;   // contiguous K layout: K = hi*16 + e
      half_t v = (half_t)0.f;
      if (K < CIN * 9) {
        int tap = K >> 1, c = K & 1;
        int dy3 = tap / 3, dx3 = tap % 3;
        v = ev_tile[((size_t)dy3 * PW + (w0 + col0 + dx3)) * CIN + c];
      }
      aI[e] = v;
    }
    const v16h b0 = *(const v16h*)&wi_lds[((0 * 16 + col0)) * 32 + hi * 16];
    const v16h b1 = *(const v16h*)&wi_lds[((1 * 16 + col0)) * 32 + hi * 16];
    const v16h b2 = *(const v16h*)&wi_lds[((2 * 16 + col0)) * 32 + hi * 16];
    const v16h b3 = *(const v16h*)&wi_lds[((3 * 16 + col0)) * 32 + hi * 16];
    acc0 = __builtin_amdgcn_wmma_f32_16x16x32_f16(false, aI, false, b0, (short)0, acc0, false, false);
    acc1 = __builtin_amdgcn_wmma_f32_16x16x32_f16(false, aI, false, b1, (short)0, acc1, false, false);
    acc2 = __builtin_amdgcn_wmma_f32_16x16x32_f16(false, aI, false, b2, (short)0, acc2, false, false);
    acc3 = __builtin_amdgcn_wmma_f32_16x16x32_f16(false, aI, false, b3, (short)0, acc3, false, false);
  }

  // recurrent conv: 9 taps, K = 32 input channels per tap
#pragma unroll
  for (int tap = 0; tap < 9; ++tap) {
    const int dy3 = tap / 3, dx3 = tap % 3;
    const half_t* abase = &sp_tile[((size_t)dy3 * PW + (w0 + col0 + dx3)) * COUT];
    v8h alo = *(const v8h*)(abase + hi * 16);        // K = hi*16 .. hi*16+7
    v8h ahi = *(const v8h*)(abase + hi * 16 + 8);    // K = hi*16+8 .. hi*16+15
    v16h aR;
#pragma unroll
    for (int e = 0; e < 8; ++e) { aR[e] = alo[e]; aR[e + 8] = ahi[e]; }
    const half_t* wb = &wg_lds[(size_t)tap * 64 * 32];
    const v16h b0 = *(const v16h*)&wb[((0 * 16 + col0)) * 32 + hi * 16];
    const v16h b1 = *(const v16h*)&wb[((1 * 16 + col0)) * 32 + hi * 16];
    const v16h b2 = *(const v16h*)&wb[((2 * 16 + col0)) * 32 + hi * 16];
    const v16h b3 = *(const v16h*)&wb[((3 * 16 + col0)) * 32 + hi * 16];
    acc0 = __builtin_amdgcn_wmma_f32_16x16x32_f16(false, aR, false, b0, (short)0, acc0, false, false);
    acc1 = __builtin_amdgcn_wmma_f32_16x16x32_f16(false, aR, false, b1, (short)0, acc1, false, false);
    acc2 = __builtin_amdgcn_wmma_f32_16x16x32_f16(false, aR, false, b2, (short)0, acc2, false, false);
    acc3 = __builtin_amdgcn_wmma_f32_16x16x32_f16(false, aR, false, b3, (short)0, acc3, false, false);
  }

  // ---- epilogue: gate, integrate, spike, soft reset, segment scatter ----
  // C/D layout: lane element r -> pixel w0 + r + 8*hi, channel-in-tile = col0
  const int pxb = w0 + hi * 8;   // this lane's 8 consecutive pixels
#pragma unroll
  for (int gidx = 0; gidx < 2; ++gidx) {
    const int ch = gidx * 16 + col0;          // gate channel in [0,32)
    v8f aG = gidx ? acc1 : acc0;              // conv channels ch
    v8f aC = gidx ? acc3 : acc2;              // conv channels 32+ch
    const size_t sidx = (((size_t)b * COUT + ch) * HH + h) * WW + pxb;
    v8f vm = *(const v8f*)&vmem[sidx];
    v8f va = *(const v8f*)&vavg[sidx];
    v8i sg = *(const v8i*)&seg[sidx];
    const float bg = bias[ch];
    const float bc = bias[32 + ch];
#pragma unroll
    for (int r = 0; r < 8; ++r) {
      float graw = aG[r] + bg;
      float cur  = aC[r] + bc;
      float gate = 1.f / (1.f + __expf(-graw));
      float v    = gate * vm[r] + cur;
      float sp   = (v > THRESH) ? 1.f : 0.f;
      vm[r] = v - THRESH * sp;
      float vnew = va[r] + v;
      int s = sg[r];
      if (sp > 0.f && s < TS) {
        agg[((((size_t)s * BATCH + b) * COUT + ch) * HH + h) * WW + pxb + r] += vnew;
        s += 1;
      }
      sg[r] = s;
      va[r] = (sp > 0.f) ? 0.f : vnew;
      spk[(((size_t)b * PW + (h + 1)) * PW + (pxb + r + 1)) * COUT + ch] = (half_t)sp;
    }
    *(v8f*)&vmem[sidx] = vm;
    *(v8f*)&vavg[sidx] = va;
    *(v8i*)&seg[sidx]  = sg;
  }
}

// ---------------- tail: flush non-spiking positions ----------------
__global__ void rsnn_final(const float* __restrict__ vavg,
                           const int*   __restrict__ seg,
                           const half_t* __restrict__ spk,
                           float* __restrict__ agg)
{
  size_t i = (size_t)blockIdx.x * blockDim.x + threadIdx.x;
  if (i >= NS) return;
  int w  = (int)(i % WW);
  int h  = (int)((i / WW) % HH);
  int ch = (int)((i / ((size_t)WW * HH)) % COUT);
  int b  = (int)(i / ((size_t)WW * HH * COUT));
  float sp = (float)spk[(((size_t)b * PW + (h + 1)) * PW + (w + 1)) * COUT + ch];
  int s = seg[i];
  if (sp == 0.f && s < TS)
    agg[((((size_t)s * BATCH + b) * COUT + ch) * HH + h) * WW + w] += vavg[i];
}

// ---------------- host entry ----------------
extern "C" void kernel_launch(void* const* d_in, const int* in_sizes, int n_in,
                              void* d_out, int out_size, void* d_ws, size_t ws_size,
                              hipStream_t stream) {
  const float* events = (const float*)d_in[0];
  const float* w_in   = (const float*)d_in[1];
  const float* b_in   = (const float*)d_in[2];
  const float* w_gate = (const float*)d_in[3];
  const float* b_gate = (const float*)d_in[4];
  float* agg = (float*)d_out;

  char* ws = (char*)d_ws;
  float*  vmem = (float*)(ws + OFF_VMEM);
  float*  vavg = (float*)(ws + OFF_VAVG);
  int*    seg  = (int*)(ws + OFF_SEG);
  half_t* spk  = (half_t*)(ws + OFF_SPK);
  half_t* wg   = (half_t*)(ws + OFF_WG);
  half_t* wi   = (half_t*)(ws + OFF_WI);
  float*  bias = (float*)(ws + OFF_BIAS);

  // zero all state (incl. padded spike borders) and the poisoned output
  rsnn_zero<<<2048, 256, 0, stream>>>((uint32_t*)ws, OFF_WG / 4);
  rsnn_zero<<<2048, 256, 0, stream>>>((uint32_t*)agg, (size_t)out_size);

  // weight conversion / layout prep
  rsnn_prep<<<(9 * 64 * 32 + 255) / 256, 256, 0, stream>>>(
      w_gate, w_in, b_in, b_gate, wg, wi, bias);

  // time-reversed recurrent scan: 16 fused steps
  for (int s = 0; s < NB; ++s) {
    rsnn_step<<<BATCH * HH, 320, 0, stream>>>(
        events, wg, wi, bias, vmem, vavg, seg, spk, agg, NB - 1 - s);
  }

  // final write for positions that did not spike on the last step
  rsnn_final<<<(int)((NS + 255) / 256), 256, 0, stream>>>(vavg, seg, spk, agg);
}
